// MCLoss_36532991820640
// MI455X (gfx1250) — matrix operand order repaired and checked by reading, not compiled, not run on hardware
//
#include <hip/hip_runtime.h>
#include <hip/hip_bf16.h>

// Problem constants (match reference)
#define BB 32
#define RR 512
#define DD 256
#define PP 5532
#define PPAD 5536           // 346 * 16
#define PTILES 346
#define NEG_INF_F (-1e30f)

// LDS staging: one proto tile = 16 rows x 512B, padded to 528B/row (bank-conflict-free)
#define ROW_LDS   528
#define TILE_LDS  (16 * ROW_LDS)

typedef __attribute__((ext_vector_type(16))) __bf16   v16bf;
typedef __attribute__((ext_vector_type(8)))  float    v8f;
typedef __attribute__((ext_vector_type(4)))  float    f32x4;
typedef __attribute__((ext_vector_type(4)))  unsigned int u32x4;

union BfVec {
    v16bf v;
    u32x4 q[2];
    unsigned short us[16];
};

__device__ __forceinline__ unsigned short f2bf(float f) {
    unsigned int u = __float_as_uint(f);
    unsigned int r = u + 0x7fffu + ((u >> 16) & 1u);   // round-to-nearest-even
    return (unsigned short)(r >> 16);
}

// gfx1250 async copy: global -> LDS, 16B per lane, tracked by ASYNCcnt
__device__ __forceinline__ void async_g2l_b128(unsigned lds_off, const void* gaddr) {
    asm volatile("global_load_async_to_lds_b128 %0, %1, off"
                 :: "v"(lds_off), "v"(gaddr) : "memory");
}

// ---------------- workspace layout (bytes) ----------------
#define OFF_SCAL  0
#define OFF_CNT   256
#define SZ_CNT    ((size_t)BB * PPAD * 4)
#define OFF_SUM   (OFF_CNT + SZ_CNT)
#define SZ_SUM    ((size_t)BB * PP * DD * 4)
#define OFF_PROTO (OFF_SUM + SZ_SUM)
#define SZ_PROTO  ((size_t)BB * PPAD * DD * 2)
#define OFF_PRES  (OFF_PROTO + SZ_PROTO)
#define ZERO_F4   ((OFF_PROTO / 4) / 4)   // float4 count covering scalars+counts+sums

// ---------------- kernel 0: zero scratch ----------------
__global__ void zero_ws_kernel(f32x4* __restrict__ p, long n4) {
    long i = (long)blockIdx.x * blockDim.x + threadIdx.x;
    long stride = (long)gridDim.x * blockDim.x;
    f32x4 z = {0.f, 0.f, 0.f, 0.f};
    for (; i < n4; i += stride) p[i] = z;
}

// ---------------- kernel 1: scatter prototypes ----------------
__global__ __launch_bounds__(256) void proto_accum_kernel(
    const float* __restrict__ inputs, const float* __restrict__ cls,
    const int* __restrict__ label, float* __restrict__ sums,
    float* __restrict__ counts, float* __restrict__ scalars)
{
    const int roi = blockIdx.x;            // 0 .. B*R-1
    const int b   = roi / RR;
    const int lab = label[roi] - 1;        // -1 = background
    if (lab < 0) return;
    const float sc = cls[roi];
    const int d = threadIdx.x;
    atomicAdd(&sums[((size_t)b * PP + lab) * DD + d],
              inputs[(size_t)roi * DD + d] * sc);
    if (d == 0) {
        atomicAdd(&counts[(size_t)b * PPAD + lab], 1.0f);
        atomicAdd(&scalars[1], 1.0f);      // n_valid
    }
}

// ---------------- kernel 2: normalize -> bf16 proto + present ----------------
__global__ __launch_bounds__(256) void proto_norm_kernel(
    const float* __restrict__ sums, const float* __restrict__ counts,
    unsigned short* __restrict__ protoBF, unsigned char* __restrict__ present)
{
    const int row = blockIdx.x;            // 0 .. B*PPAD-1
    const int b = row / PPAD;
    const int p = row - b * PPAD;
    const int tid = threadIdx.x;

    float v = 0.f;
    if (p < PP) v = sums[((size_t)b * PP + p) * DD + tid];

    __shared__ float sh[256];
    sh[tid] = v * v;
    __syncthreads();
    for (int o = 128; o > 0; o >>= 1) {
        if (tid < o) sh[tid] += sh[tid + o];
        __syncthreads();
    }
    const float nrm = sqrtf(sh[0]);
    // proto = sums/max(c,1) then /max(norm,1e-12)  ==  sums/||sums||  (scaling cancels)
    const float inv = (nrm > 1e-12f) ? (1.0f / nrm) : 0.0f;
    protoBF[(size_t)row * DD + tid] = f2bf(v * inv);
    if (tid == 0)
        present[row] = (p < PP && counts[row] > 0.0f) ? 1 : 0;
}

// ---------------- kernel 3: fused WMMA GEMM + masked online softmax + NLL ----
// 4 waves per block; all 4 waves are in the SAME image, so the proto tile is
// staged ONCE per block into LDS via async global->LDS copies (ASYNCcnt),
// double-buffered so tile pt+1 streams while WMMAs consume tile pt.
__global__ __launch_bounds__(128) void gemm_softmax_nll_kernel(
    const float* __restrict__ inputs, const float* __restrict__ cls,
    const int* __restrict__ label, const unsigned short* __restrict__ protoBF,
    const unsigned char* __restrict__ present, float* __restrict__ scalars)
{
    __shared__ __align__(16) char smem[2 * TILE_LDS];

    const int wid  = threadIdx.x >> 5;
    const int lane = threadIdx.x & 31;
    const int tile = blockIdx.x * 4 + wid;       // 0 .. B*R/16 - 1
    const int b     = tile >> 5;                 // 32 tiles per image (same b for whole block)
    const int rbase = (tile & 31) << 4;
    const int rloc  = lane & 15;
    const int hi    = lane >> 4;                 // 0: lanes 0-15, 1: lanes 16-31
    const int roi   = b * RR + rbase + rloc;

    // ---- build B operand (x^T tile, scaled, bf16) once; lives in VGPRs ----
    // B layout: lane = N (ROI), lanes<16 hold K0..15, lanes>=16 hold K16..31.
    const float cs = cls[roi];
    const float* xrow = inputs + (size_t)roi * DD;
    BfVec breg[8];
#pragma unroll
    for (int c = 0; c < 8; ++c) {
        const int kb = c * 32 + hi * 16;
#pragma unroll
        for (int j = 0; j < 4; ++j) {
            f32x4 f = *(const f32x4*)(xrow + kb + 4 * j);
            breg[c].us[4 * j + 0] = f2bf(f.x * cs);
            breg[c].us[4 * j + 1] = f2bf(f.y * cs);
            breg[c].us[4 * j + 2] = f2bf(f.z * cs);
            breg[c].us[4 * j + 3] = f2bf(f.w * cs);
        }
    }

    const int lr = label[roi] - 1;
    const bool validr = (lr >= 0);
    const int lbl = validr ? lr : 0;

    float m = NEG_INF_F, s = 0.f, tv = NEG_INF_F;  // per-lane online softmax state
    const char* aimg = (const char*)protoBF + (size_t)b * PPAD * DD * 2;
    const unsigned char* presb = present + (size_t)b * PPAD;

    // LDS byte offsets (ISA: generic LDS address truncated to 32 bits = LDS offset)
    const unsigned smem_off = (unsigned)(size_t)(void*)smem;

    // stage proto tile pt into LDS buffer bufIdx: 512 x 16B chunks, 4 per thread
    auto stage = [&](int pt, int bufIdx) {
        const char* gbase = aimg + (size_t)(pt << 4) * (DD * 2);
#pragma unroll
        for (int rnd = 0; rnd < 4; ++rnd) {
            const int q   = threadIdx.x + rnd * 128;   // 0..511
            const int row = q >> 5;
            const int col = q & 31;
            async_g2l_b128(smem_off + bufIdx * TILE_LDS + row * ROW_LDS + col * 16,
                           gbase + row * 512 + col * 16);
        }
    };

    stage(0, 0);   // prologue: each wave now has 4 async ops in flight

    for (int pt = 0; pt < PTILES; ++pt) {
        if (pt + 1 < PTILES) {
            stage(pt + 1, (pt + 1) & 1);
            asm volatile("s_wait_asynccnt 0x4" ::: "memory");  // tile pt landed
        } else {
            asm volatile("s_wait_asynccnt 0x0" ::: "memory");
        }
        __syncthreads();   // all waves' async writes of tile pt visible

        const int pbase = pt << 4;
        // A operand from LDS: lane = M (pid); lanes<16: K0..15; lanes>=16: K16..31.
        const char* arow = smem + (pt & 1) * TILE_LDS + rloc * ROW_LDS;
        BfVec a[8];
#pragma unroll
        for (int c = 0; c < 8; ++c) {
            const int off = c * 64 + hi * 32;
            a[c].q[0] = *(const u32x4*)(arow + off);
            a[c].q[1] = *(const u32x4*)(arow + off + 16);
        }
        v8f acc = {0.f, 0.f, 0.f, 0.f, 0.f, 0.f, 0.f, 0.f};
#pragma unroll
        for (int c = 0; c < 8; ++c)
            acc = __builtin_amdgcn_wmma_f32_16x16x32_bf16(
                false, a[c].v, false, breg[c].v, (short)0, acc, false, false);

        // present mask: rows are lane-half uniform -> one u64 of 8 bytes
        const unsigned long long p8 =
            *(const unsigned long long*)(presb + pbase + hi * 8);

        float lv[8];
        float tmax = NEG_INF_F;
#pragma unroll
        for (int v = 0; v < 8; ++v) {
            const int prow = pbase + hi * 8 + v;
            float lg = ((p8 >> (8 * v)) & 0xffULL) ? acc[v] : NEG_INF_F;
            if (prow == lbl) tv = lg;          // target logit capture
            lv[v] = lg;
            tmax = fmaxf(tmax, lg);
        }
        const float nm = fmaxf(m, tmax);
        if (nm > -5e29f) {                      // skip while everything masked
            const float rescale = __expf(m - nm);
            float acc_s = 0.f;
#pragma unroll
            for (int v = 0; v < 8; ++v) acc_s += __expf(lv[v] - nm);
            s = s * rescale + acc_s;
            m = nm;
        }

        __syncthreads();   // protect buffer (pt & 1) before it is re-staged
    }

    // merge the two lane-halves of each ROI (lane j <-> lane j+16)
    const float om = __shfl_xor(m, 16);
    const float os = __shfl_xor(s, 16);
    const float ot = __shfl_xor(tv, 16);
    const float M = fmaxf(m, om);
    const float S = ((m  > -5e29f) ? s  * __expf(m  - M) : 0.f) +
                    ((om > -5e29f) ? os * __expf(om - M) : 0.f);
    const float T = fmaxf(tv, ot);

    if (lane < 16) {
        const float nll = validr ? -(T - M - __logf(S)) : 0.f;
        atomicAdd(&scalars[0], nll);
    }
}

// ---------------- kernel 4: finalize ----------------
__global__ void finalize_kernel(const float* __restrict__ scalars,
                                float* __restrict__ out)
{
    if (threadIdx.x == 0 && blockIdx.x == 0)
        out[0] = scalars[0] / fmaxf(scalars[1], 1.0f);
}

extern "C" void kernel_launch(void* const* d_in, const int* in_sizes, int n_in,
                              void* d_out, int out_size, void* d_ws, size_t ws_size,
                              hipStream_t stream) {
    (void)in_sizes; (void)n_in; (void)out_size; (void)ws_size;

    const float* inputs = (const float*)d_in[0];   // [B*R, D] f32
    const float* cls    = (const float*)d_in[1];   // [B*R, 1] f32
    const int*   label  = (const int*)d_in[2];     // [B, R]  int32

    char* ws = (char*)d_ws;
    float*          scalars = (float*)(ws + OFF_SCAL);
    float*          counts  = (float*)(ws + OFF_CNT);
    float*          sums    = (float*)(ws + OFF_SUM);
    unsigned short* protoBF = (unsigned short*)(ws + OFF_PROTO);
    unsigned char*  present = (unsigned char*)(ws + OFF_PRES);

    // 0) zero scalars + counts + sums
    zero_ws_kernel<<<8192, 256, 0, stream>>>((f32x4*)ws, (long)ZERO_F4);

    // 1) scatter-add prototypes (one block per ROI)
    proto_accum_kernel<<<BB * RR, 256, 0, stream>>>(
        inputs, cls, label, sums, counts, scalars);

    // 2) normalize rows, convert to bf16, mark present (one block per padded row)
    proto_norm_kernel<<<BB * PPAD, 256, 0, stream>>>(
        sums, counts, protoBF, present);

    // 3) fused WMMA GEMM + online softmax + NLL  (1024 wave-tiles, 4/block)
    gemm_softmax_nll_kernel<<<(BB * RR / 16) / 4, 128, 0, stream>>>(
        inputs, cls, label, protoBF, present, scalars);

    // 4) finalize
    finalize_kernel<<<1, 32, 0, stream>>>(scalars, (float*)d_out);
}
